// MultiHeadAttention_24541443129426
// MI455X (gfx1250) — matrix operand order repaired and checked by reading, not compiled, not run on hardware
//
#include <hip/hip_runtime.h>

// ---------------------------------------------------------------- types
typedef __bf16 bf16;
typedef __attribute__((ext_vector_type(4)))  bf16  v4bf;
typedef __attribute__((ext_vector_type(8)))  bf16  v8bf;
typedef __attribute__((ext_vector_type(16))) bf16  v16bf;
typedef __attribute__((ext_vector_type(8)))  float v8f;
typedef __attribute__((ext_vector_type(4)))  unsigned int v4u;
typedef __attribute__((ext_vector_type(8)))  unsigned int v8u;

#define DEV __device__ __forceinline__

static constexpr int Bb = 4;
static constexpr int Ss = 4096;
static constexpr int Dd = 1024;
static constexpr float SCALE = 0.125f;   // 1/sqrt(64)

struct BfPair { v8bf lo, hi; };

DEV v16bf make_frag(v8bf lo, v8bf hi) {
  BfPair p{lo, hi};
  return __builtin_bit_cast(v16bf, p);
}

DEV v8f zero8() {
  v8f z;
#pragma unroll
  for (int i = 0; i < 8; i++) z[i] = 0.0f;
  return z;
}

// truncate a generic (flat) pointer to a shared variable to its 32-bit LDS offset
DEV unsigned lds_off32(const void* p) {
  return (unsigned)(unsigned long long)p;
}

// one-lane 16B async copy: global -> LDS (tracked by ASYNCcnt)
DEV void async_b128(unsigned lds_byte_off, const void* g) {
  asm volatile("global_load_async_to_lds_b128 %0, %1, off"
               :: "v"(lds_byte_off), "v"((unsigned long long)g)
               : "memory");
}

DEV void wait_async0() {
  asm volatile("s_wait_asynccnt 0x0" ::: "memory");
}

// TDM: 2-D tile load global -> LDS (tracked by TENSORcnt).
// data_size = 2 bytes.  Dims/strides in elements.  Optional LDS padding:
// pad_interval code p -> pad after every 2^(p+1) DWORDs; pad_amount code a -> (a+1) DWORDs.
DEV void tdm_load_2d(unsigned lds_byte_off, const void* gaddr,
                     unsigned tile_d0, unsigned tile_d1,
                     unsigned tensor_d0, unsigned tensor_d1,
                     unsigned stride0,
                     unsigned pad_interval_code, unsigned pad_amount_code,
                     bool pad_en) {
  unsigned long long ga = (unsigned long long)gaddr;
  v4u g0;
  g0[0] = 1u;                                            // count=1, user mode
  g0[1] = lds_byte_off;                                  // lds_addr
  g0[2] = (unsigned)(ga & 0xffffffffu);                  // global_addr[31:0]
  g0[3] = (unsigned)((ga >> 32) & 0x01ffffffu) | (2u << 30); // addr[56:32], type=2
  v8u g1;
  g1[0] = (1u << 16)                                     // data_size = 2 bytes
        | (pad_en ? (1u << 20) : 0u)
        | (pad_interval_code << 22)
        | (pad_amount_code << 25);
  g1[1] = (tensor_d0 & 0xffffu) << 16;                   // tensor_dim0[15:0]
  g1[2] = ((tensor_d0 >> 16) & 0xffffu) | ((tensor_d1 & 0xffffu) << 16);
  g1[3] = ((tensor_d1 >> 16) & 0xffffu) | ((tile_d0 & 0xffffu) << 16);
  g1[4] = (tile_d1 & 0xffffu);                           // tile_dim1, tile_dim2=0
  g1[5] = stride0;                                       // tensor_dim0_stride[31:0]
  g1[6] = 0u;
  g1[7] = 0u;
  asm volatile("tensor_load_to_lds %0, %1" :: "s"(g0), "s"(g1) : "memory");
}

// A-matrix 16x32 bf16 fragment (ISA 7.12.2): lanes 0-15 row M=lane, K 0-7 & 16-23;
// lanes 16-31 row M=lane-16, K 8-15 & 24-31.
DEV v16bf ld_a_frag(const bf16* __restrict__ base, int lane, int ld) {
  const int m = lane & 15;
  const int h = (lane >> 4) << 3;
  const bf16* r = base + m * ld;
  v8bf lo = *(const v8bf*)(r + h);
  v8bf hi = *(const v8bf*)(r + 16 + h);
  return make_frag(lo, hi);
}

// B-matrix 32x16 bf16 fragment: lane n<16 -> column n, K 0-7 & 16-23; lane n+16 -> K 8-15 & 24-31.
// LDS holds B transposed as [n][k], stride ld.
DEV v16bf ld_b_frag(const bf16* __restrict__ base, int lane, int ld) {
  const int n = lane & 15;
  const int h = (lane >> 4) << 3;
  const bf16* r = base + n * ld + h;
  v8bf lo = *(const v8bf*)(r);
  v8bf hi = *(const v8bf*)(r + 16);
  return make_frag(lo, hi);
}

DEV v8f wmma_bf16(v16bf a, v16bf b, v8f c) {
  return __builtin_amdgcn_wmma_f32_16x16x32_bf16(false, a, false, b, (short)0, c,
                                                 false, false);
}

// ---------------------------------------------------------------- GEMM core
// C[M,N] = A[M,K] * B^T(stored [N][K] bf16) + bias.  BM=128 BN=128 BK=32, 256 thr,
// wave grid 2(M)x4(N): each wave 64x32 = 4x2 tiles.  B tile (and bf16 A tile)
// streamed with global_load_async_to_lds_b128; fp32 A tile converted through regs.
template <bool A_F32, bool OUT_F32>
DEV void gemm_core(const void* __restrict__ Ag_, const bf16* __restrict__ Bt,
                   const float* __restrict__ bias, void* __restrict__ Cg_,
                   int M, int N, int K) {
  __shared__ bf16 a_s[2][128 * 40];
  __shared__ bf16 b_s[2][128 * 40];

  const int tid  = threadIdx.x;
  const int lane = tid & 31;
  const int wid  = tid >> 5;
  const int wm   = wid & 1;
  const int wn   = wid >> 1;
  const int m0   = blockIdx.y * 128;
  const int n0   = blockIdx.x * 128;

  v8f acc[4][2];
#pragma unroll
  for (int i = 0; i < 4; i++)
#pragma unroll
    for (int j = 0; j < 2; j++) acc[i][j] = zero8();

  float4 aF[4];

  auto b_async = [&](int kt, int buf) {
    const int k0 = kt * 32;
    const unsigned base = lds_off32(&b_s[buf][0]);
#pragma unroll
    for (int j = 0; j < 2; j++) {
      int i = j * 256 + tid, r = i >> 2, c = i & 3;
      async_b128(base + (unsigned)(r * 40 + c * 8) * 2,
                 Bt + (size_t)(n0 + r) * K + k0 + c * 8);
    }
  };

  auto a_async = [&](int kt, int buf) {
    const bf16* Ag = (const bf16*)Ag_;
    const int k0 = kt * 32;
    const unsigned base = lds_off32(&a_s[buf][0]);
#pragma unroll
    for (int j = 0; j < 2; j++) {
      int i = j * 256 + tid, r = i >> 2, c = i & 3;
      async_b128(base + (unsigned)(r * 40 + c * 8) * 2,
                 Ag + (size_t)(m0 + r) * K + k0 + c * 8);
    }
  };

  auto a_gload = [&](int kt) {
    const float* Ag = (const float*)Ag_;
    const int k0 = kt * 32;
#pragma unroll
    for (int j = 0; j < 4; j++) {
      int i = j * 256 + tid, r = i >> 3, c = i & 7;
      aF[j] = *(const float4*)(Ag + (size_t)(m0 + r) * K + k0 + c * 4);
    }
  };

  auto a_sstore = [&](int buf) {
#pragma unroll
    for (int j = 0; j < 4; j++) {
      int i = j * 256 + tid, r = i >> 3, c = i & 7;
      v4bf t;
      t[0] = (bf16)aF[j].x; t[1] = (bf16)aF[j].y;
      t[2] = (bf16)aF[j].z; t[3] = (bf16)aF[j].w;
      *(v4bf*)&a_s[buf][r * 40 + c * 4] = t;
    }
  };

  const int steps = K / 32;
  b_async(0, 0);
  if constexpr (A_F32) { a_gload(0); a_sstore(0); }
  else                 { a_async(0, 0); }
  wait_async0();
  __syncthreads();

  for (int kt = 0; kt < steps; kt++) {
    const int buf = kt & 1;
    if (kt + 1 < steps) {
      b_async(kt + 1, buf ^ 1);
      if constexpr (A_F32) a_gload(kt + 1);
      else                 a_async(kt + 1, buf ^ 1);
    }
    v16bf af[4], bfr[2];
#pragma unroll
    for (int mi = 0; mi < 4; mi++)
      af[mi] = ld_a_frag(&a_s[buf][(wm * 64 + mi * 16) * 40], lane, 40);
#pragma unroll
    for (int ni = 0; ni < 2; ni++)
      bfr[ni] = ld_b_frag(&b_s[buf][(wn * 32 + ni * 16) * 40], lane, 40);
#pragma unroll
    for (int mi = 0; mi < 4; mi++)
#pragma unroll
      for (int ni = 0; ni < 2; ni++)
        acc[mi][ni] = wmma_bf16(af[mi], bfr[ni], acc[mi][ni]);
    if (A_F32 && kt + 1 < steps) a_sstore(buf ^ 1);
    wait_async0();
    __syncthreads();
  }

  const int nl = lane & 15;
  const int h8 = (lane >> 4) << 3;
#pragma unroll
  for (int mi = 0; mi < 4; mi++) {
#pragma unroll
    for (int ni = 0; ni < 2; ni++) {
      const int   gc  = n0 + wn * 32 + ni * 16 + nl;
      const float bv  = bias[gc];
      const int   gr0 = m0 + wm * 64 + mi * 16 + h8;
#pragma unroll
      for (int i = 0; i < 8; i++) {
        float v = acc[mi][ni][i] + bv;
        if constexpr (OUT_F32)
          ((float*)Cg_)[(size_t)(gr0 + i) * N + gc] = v;
        else
          ((bf16*)Cg_)[(size_t)(gr0 + i) * N + gc] = (bf16)v;
      }
    }
  }
}

// ---------------------------------------------------------------- kernels
__global__ __launch_bounds__(256) void cvt_w_kernel(const float* __restrict__ W,
                                                    bf16* __restrict__ Wt) {
  int idx = blockIdx.x * 256 + threadIdx.x;
  int n = idx >> 10;
  int k = idx & 1023;
  Wt[idx] = (bf16)W[(size_t)k * Dd + n];   // Wt[out][in]
}

__global__ __launch_bounds__(256) void proj_kernel(
    const float* __restrict__ Q, const float* __restrict__ K,
    const float* __restrict__ V, const bf16* __restrict__ Wqt,
    const bf16* __restrict__ Wkt, const bf16* __restrict__ Wvt,
    const float* __restrict__ bq, const float* __restrict__ bk,
    const float* __restrict__ bv, bf16* __restrict__ qp, bf16* __restrict__ kp,
    bf16* __restrict__ vp) {
  const float* A; const bf16* Bt; const float* bias; bf16* C;
  if (blockIdx.z == 0)      { A = Q; Bt = Wqt; bias = bq; C = qp; }
  else if (blockIdx.z == 1) { A = K; Bt = Wkt; bias = bk; C = kp; }
  else                      { A = V; Bt = Wvt; bias = bv; C = vp; }
  gemm_core<true, false>(A, Bt, bias, C, Bb * Ss, Dd, Dd);
}

__global__ __launch_bounds__(256) void oproj_kernel(const bf16* __restrict__ attn,
                                                    const bf16* __restrict__ Wot,
                                                    const float* __restrict__ bo,
                                                    float* __restrict__ out) {
  gemm_core<false, true>(attn, Wot, bo, out, Bb * Ss, Dd, Dd);
}

// Flash attention: WG = 32 queries of one batch.  8 waves / 256 threads.
// Q and K tiles loaded by the Tensor Data Mover (with LDS pad: +16B per 1024B,
// i.e. rows at 2080B stride with a +16B skew at d=512).  V transposed manually.
__global__ __launch_bounds__(256) void flash_kernel(const bf16* __restrict__ qp,
                                                    const bf16* __restrict__ kp,
                                                    const bf16* __restrict__ vp,
                                                    bf16* __restrict__ attn) {
  constexpr int QT = 32, KB = 32;
  constexpr int LROW = 1040;   // elements per padded row (2080 B)
  constexpr int LV = 40;

  __shared__ bf16  q_s[QT * LROW];      //  66,560 B
  __shared__ bf16  kv_s[Dd * LV];       //  81,920 B : K [32][LROW] then V^T [1024][LV]
  __shared__ float sc_s[2][QT][36];     //   9,216 B
  __shared__ bf16  p_s[QT][40];         //   2,560 B
  __shared__ float m_s[QT], l_s[QT], f_s[QT];

  const int tid  = threadIdx.x;
  const int lane = tid & 31;
  const int wid  = tid >> 5;
  const int q0   = blockIdx.x * QT;
  const int bb   = blockIdx.y;

  // Q tile via TDM (one wave issues the whole 32x1024 tile)
  if (wid == 0) {
    const bf16* qg = qp + ((size_t)bb * Ss + q0) * Dd;
    tdm_load_2d(lds_off32(&q_s[0]), qg,
                /*tile*/ Dd, QT, /*tensor*/ Dd, QT, /*stride0*/ Dd,
                /*pad: every 256 dwords add 4 dwords*/ 7u, 3u, true);
  }
  if (tid < QT) { m_s[tid] = -1.0e30f; l_s[tid] = 0.0f; }

  v8f oacc[2][8];
#pragma unroll
  for (int a = 0; a < 2; a++)
#pragma unroll
    for (int b = 0; b < 8; b++) oacc[a][b] = zero8();

  if (wid == 0) __builtin_amdgcn_s_wait_tensorcnt(0);
  __syncthreads();

  const int nblocks = q0 / KB + 1;
  const int mt = wid & 1, nt = (wid >> 1) & 1, dh = wid >> 2;
  const int h8 = (lane >> 4) << 3;
  const int nbase = wid * 128;

  for (int kb = 0; kb < nblocks; kb++) {
    const int k0 = kb * KB;

    // ---- K block via TDM -> kv_s as [KB][LROW] (same pad scheme)
    if (wid == 0) {
      const bf16* kg = kp + ((size_t)bb * Ss + k0) * Dd;
      tdm_load_2d(lds_off32(&kv_s[0]), kg,
                  Dd, KB, Dd, KB, Dd, 7u, 3u, true);
      __builtin_amdgcn_s_wait_tensorcnt(0);
    }
    __syncthreads();

    // ---- scores: each wave one 16x16 tile over half of D
    {
      v8f s = zero8();
      for (int dd = 0; dd < 16; dd++) {
        const int d   = dh * 512 + dd * 32;
        const int dsk = d + (d >= 512 ? 8 : 0);     // fold TDM pad skew
        v16bf a = ld_a_frag(q_s + (mt * 16) * LROW + dsk, lane, LROW);
        v16bf b = ld_b_frag(kv_s + (nt * 16) * LROW + dsk, lane, LROW);
        s = wmma_bf16(a, b, s);
      }
      const int col = nt * 16 + (lane & 15);
#pragma unroll
      for (int i = 0; i < 8; i++)
        sc_s[dh][mt * 16 + i + h8][col] = s[i];
    }
    __syncthreads();

    // ---- V block -> kv_s transposed: [dim][key]
    {
      const bf16* vg = vp + ((size_t)bb * Ss + k0) * Dd;
#pragma unroll
      for (int j = 0; j < 16; j++) {
        int i = j * 256 + tid;
        int r = i >> 7;
        int c = (i & 127) * 8;
        v8bf t = *(const v8bf*)(vg + (size_t)r * Dd + c);
#pragma unroll
        for (int e = 0; e < 8; e++) kv_s[(c + e) * LV + r] = t[e];
      }
    }

    // ---- online softmax: one lane per query row
    if (tid < QT) {
      const int   row = tid;
      const int   qgl = q0 + row;
      const float mo  = m_s[row];
      float mx = mo;
      for (int c = 0; c < KB; c++) {
        float v = (k0 + c <= qgl)
                      ? (sc_s[0][row][c] + sc_s[1][row][c]) * SCALE
                      : -3.0e38f;
        mx = fmaxf(mx, v);
      }
      float sum = 0.0f;
      for (int c = 0; c < KB; c++) {
        bool  ok = (k0 + c <= qgl);
        float v  = ok ? (sc_s[0][row][c] + sc_s[1][row][c]) * SCALE : 0.0f;
        float p  = ok ? __expf(v - mx) : 0.0f;
        p_s[row][c] = (bf16)p;
        sum += p;
      }
      const float f = __expf(mo - mx);
      m_s[row] = mx;
      l_s[row] = l_s[row] * f + sum;
      f_s[row] = f;
    }
    __syncthreads();

    // ---- rescale accumulators
#pragma unroll
    for (int m2 = 0; m2 < 2; m2++) {
      float fr[8];
#pragma unroll
      for (int i = 0; i < 8; i++) fr[i] = f_s[m2 * 16 + i + h8];
#pragma unroll
      for (int n2 = 0; n2 < 8; n2++)
#pragma unroll
        for (int i = 0; i < 8; i++) oacc[m2][n2][i] *= fr[i];
    }

    // ---- O += P * V
#pragma unroll
    for (int m2 = 0; m2 < 2; m2++) {
      v16bf a = ld_a_frag(&p_s[m2 * 16][0], lane, 40);
#pragma unroll
      for (int n2 = 0; n2 < 8; n2++) {
        v16bf b = ld_b_frag(kv_s + (nbase + n2 * 16) * LV, lane, LV);
        oacc[m2][n2] = wmma_bf16(a, b, oacc[m2][n2]);
      }
    }
    __syncthreads();
  }

  // ---- finalize
  {
    bf16* og = attn + ((size_t)bb * Ss + q0) * Dd;
    const int nl = lane & 15;
#pragma unroll
    for (int m2 = 0; m2 < 2; m2++) {
      float inv[8];
#pragma unroll
      for (int i = 0; i < 8; i++) inv[i] = 1.0f / l_s[m2 * 16 + i + h8];
#pragma unroll
      for (int n2 = 0; n2 < 8; n2++) {
        const int gc = nbase + n2 * 16 + nl;
#pragma unroll
        for (int i = 0; i < 8; i++)
          og[(size_t)(m2 * 16 + i + h8) * Dd + gc] =
              (bf16)(oacc[m2][n2][i] * inv[i]);
      }
    }
  }
}

// ---------------------------------------------------------------- launch
extern "C" void kernel_launch(void* const* d_in, const int* in_sizes, int n_in,
                              void* d_out, int out_size, void* d_ws,
                              size_t ws_size, hipStream_t stream) {
  (void)in_sizes; (void)n_in; (void)out_size; (void)ws_size;

  const float* Q  = (const float*)d_in[0];
  const float* Ki = (const float*)d_in[1];
  const float* Vi = (const float*)d_in[2];
  const float* Wq = (const float*)d_in[3];
  const float* bq = (const float*)d_in[4];
  const float* Wk = (const float*)d_in[5];
  const float* bk = (const float*)d_in[6];
  const float* Wv = (const float*)d_in[7];
  const float* bv = (const float*)d_in[8];
  const float* Wo = (const float*)d_in[9];
  const float* bo = (const float*)d_in[10];
  float* out = (float*)d_out;

  const size_t BSD  = (size_t)Bb * Ss * Dd;
  const size_t DDsz = (size_t)Dd * Dd;
  bf16* qp   = (bf16*)d_ws;
  bf16* kp   = qp + BSD;
  bf16* vp   = kp + BSD;
  bf16* attn = vp + BSD;
  bf16* Wqt  = attn + BSD;
  bf16* Wkt  = Wqt + DDsz;
  bf16* Wvt  = Wkt + DDsz;
  bf16* Wot  = Wvt + DDsz;

  const dim3 thr(256);
  const unsigned cvtBlocks = (unsigned)(DDsz / 256);
  cvt_w_kernel<<<dim3(cvtBlocks), thr, 0, stream>>>(Wq, Wqt);
  cvt_w_kernel<<<dim3(cvtBlocks), thr, 0, stream>>>(Wk, Wkt);
  cvt_w_kernel<<<dim3(cvtBlocks), thr, 0, stream>>>(Wv, Wvt);
  cvt_w_kernel<<<dim3(cvtBlocks), thr, 0, stream>>>(Wo, Wot);

  proj_kernel<<<dim3(Dd / 128, (Bb * Ss) / 128, 3), thr, 0, stream>>>(
      Q, Ki, Vi, Wqt, Wkt, Wvt, bq, bk, bv, qp, kp, vp);

  flash_kernel<<<dim3(Ss / 32, Bb), thr, 0, stream>>>(qp, kp, vp, attn);

  oproj_kernel<<<dim3(Dd / 128, (Bb * Ss) / 128), thr, 0, stream>>>(attn, Wot,
                                                                    bo, out);
}
